// RD_NOISING_44289702756646
// MI455X (gfx1250) — matrix operand order repaired and checked by reading, not compiled, not run on hardware
//
#include <hip/hip_runtime.h>
#include <hip/hip_bf16.h>
#include <math.h>

// ---- problem constants (from reference setup_inputs) ----
#define BB 8
#define CC 2048
#define HH 28
#define WW 28
#define HWW (HH * WW)           // 784
#define NQ (BB * HWW)           // 6272 query rows
#define MM 8192                 // memory rows
#define TOPK 9
#define NSLICE 8
#define SLICE_M (MM / NSLICE)   // 1024
#define NROWT (NQ / 16)         // 392 16-row tiles
#define EPSV 1e-8f

// K4 blocking
#define QTILE 64                // query rows per workgroup (4 waves x 16)
#define CTILES 4                // column tiles per pass (64 memory rows)
#define KBLK 128                // K block staged in LDS
#define BPAD 8                  // row padding (shorts) to avoid LDS bank conflicts
#define BSTR (KBLK + BPAD)      // 136 shorts per staged row

typedef __attribute__((ext_vector_type(8)))  float  v8f;
typedef __attribute__((ext_vector_type(8)))  __bf16 bf8v;
typedef __attribute__((ext_vector_type(16))) __bf16 bf16v;

#define SHUF16(a, b) __builtin_shufflevector((a), (b), 0,1,2,3,4,5,6,7,8,9,10,11,12,13,14,15)
#define WMMA_BF16(A, B, C) __builtin_amdgcn_wmma_f32_16x16x32_bf16(false, (A), false, (B), (short)0, (C), false, false)

// round-to-nearest-even f32 -> bf16, plus bf16 residual
__device__ inline void splitbf(float x, unsigned short& hi, unsigned short& lo) {
    unsigned int u  = __float_as_uint(x);
    unsigned int r  = u + 0x7FFFu + ((u >> 16) & 1u);
    unsigned short h = (unsigned short)(r >> 16);
    float hf = __uint_as_float(((unsigned int)h) << 16);
    float l  = x - hf;
    unsigned int ul = __float_as_uint(l);
    unsigned int rl = ul + 0x7FFFu + ((ul >> 16) & 1u);
    hi = h;
    lo = (unsigned short)(rl >> 16);
}

// ---------------- K1: elementwise split of memory bank ----------------
__global__ __launch_bounds__(256) void k_split(const float* __restrict__ src,
                                               unsigned short* __restrict__ hi,
                                               unsigned short* __restrict__ lo,
                                               int n) {
    int i = blockIdx.x * 256 + threadIdx.x;
    if (i >= n) return;
    unsigned short h, l;
    splitbf(src[i], h, l);
    hi[i] = h; lo[i] = l;
}

// ---------------- K2: per-row squared norms ----------------
__global__ __launch_bounds__(256) void k_rownorm2(const float* __restrict__ src,
                                                  float* __restrict__ out) {
    int row = blockIdx.x;
    const float* p = src + (size_t)row * CC;
    float s = 0.f;
    for (int c = threadIdx.x; c < CC; c += 256) { float x = p[c]; s += x * x; }
    __shared__ float red[256];
    red[threadIdx.x] = s; __syncthreads();
    for (int st = 128; st > 0; st >>= 1) {
        if (threadIdx.x < st) red[threadIdx.x] += red[threadIdx.x + st];
        __syncthreads();
    }
    if (threadIdx.x == 0) out[row] = red[0];
}

// ---------------- K3: NCHW -> NHWC transposed bf16 split of features + ||f||^2 ----------------
__global__ __launch_bounds__(256) void k_convf(const float* __restrict__ feat,
                                               unsigned short* __restrict__ fhi,
                                               unsigned short* __restrict__ flo,
                                               float* __restrict__ f2) {
    int n = blockIdx.x;
    int b = n / HWW; int rem = n % HWW;
    size_t hw = (size_t)rem;                 // h*W + w
    float s = 0.f;
    for (int c = threadIdx.x; c < CC; c += 256) {
        float x = feat[((size_t)b * CC + c) * HWW + hw];
        unsigned short h, l;
        splitbf(x, h, l);
        fhi[(size_t)n * CC + c] = h;
        flo[(size_t)n * CC + c] = l;
        s += x * x;
    }
    __shared__ float red[256];
    red[threadIdx.x] = s; __syncthreads();
    for (int st = 128; st > 0; st >>= 1) {
        if (threadIdx.x < st) red[threadIdx.x] += red[threadIdx.x + st];
        __syncthreads();
    }
    if (threadIdx.x == 0) f2[n] = red[0];
}

// ---------------- K4: fused bf16x3 WMMA GEMM + partial top-9 ----------------
// grid = (NQ/QTILE, NSLICE), block = 128 (4 waves).
// Workgroup: 64 query rows x 1024 memory rows. Wave w owns query rows [qBase+16w, +16).
// B tiles (64 mem rows x KBLK, hi+lo) staged in LDS with async-to-LDS, shared by all 4 waves.
// Each wave register-blocks 4 column tiles (4 x v8f accumulators) so A fragments are reused 4x.
__global__ __launch_bounds__(128) void k_gemm_topk(const unsigned short* __restrict__ fhi,
                                                   const unsigned short* __restrict__ flo,
                                                   const unsigned short* __restrict__ mhi,
                                                   const unsigned short* __restrict__ mlo,
                                                   const float* __restrict__ f2,
                                                   const float* __restrict__ m2,
                                                   float* __restrict__ pkd,
                                                   int* __restrict__ pki) {
    const int tid   = threadIdx.x;
    const int wv    = tid >> 5;        // wave id 0..3
    const int lane  = tid & 31;
    const int hl    = lane >> 4;       // half-wave
    const int ln    = lane & 15;
    const int qBase = blockIdx.x * QTILE + wv * 16;
    const int mBase = blockIdx.y * SLICE_M;

    __shared__ __align__(16) unsigned short sBhi[CTILES * 16][BSTR];
    __shared__ __align__(16) unsigned short sBlo[CTILES * 16][BSTR];
    __shared__ float sDist[4][CTILES][16][16];

    float bestd[TOPK]; int besti[TOPK];
#pragma unroll
    for (int j = 0; j < TOPK; ++j) { bestd[j] = 3.0e38f; besti[j] = 0; }

    float f2v[8];
#pragma unroll
    for (int r = 0; r < 8; ++r) f2v[r] = f2[qBase + r + hl * 8];

    // A fragment source row (16x32 bf16 layout: M = lane&15)
    const unsigned short* fhiRow = fhi + (size_t)(qBase + ln) * CC;
    const unsigned short* floRow = flo + (size_t)(qBase + ln) * CC;

    for (int g = 0; g < SLICE_M / (CTILES * 16); ++g) {
        const int colBase = mBase + g * (CTILES * 16);

        v8f acc0 = {}, acc1 = {}, acc2 = {}, acc3 = {};

        for (int kb = 0; kb < CC; kb += KBLK) {
            // ---- stage B block: 64 rows x KBLK (hi then lo), 16B chunks, async to LDS ----
            // 2048 chunks total; thread handles chunk = tid + i*128; i<8 -> hi, i>=8 -> lo.
#pragma unroll
            for (int i = 0; i < 16; ++i) {
                const int rem = tid + (i & 7) * 128;     // 0..1023
                const int row = rem >> 4;                // 0..63
                const int seg = rem & 15;                // 16B chunk within row
                const unsigned short* gsrc;
                const unsigned short* ldst;
                if (i < 8) {
                    gsrc = mhi + (size_t)(colBase + row) * CC + kb + seg * 8;
                    ldst = &sBhi[row][seg * 8];
                } else {
                    gsrc = mlo + (size_t)(colBase + row) * CC + kb + seg * 8;
                    ldst = &sBlo[row][seg * 8];
                }
                unsigned lds_addr = (unsigned)(uintptr_t)ldst;   // low 32 bits = LDS offset
                unsigned long long ga = (unsigned long long)(uintptr_t)gsrc;
                asm volatile("global_load_async_to_lds_b128 %0, %1, off"
                             :: "v"(lds_addr), "v"(ga) : "memory");
            }
            asm volatile("s_wait_asynccnt 0x0" ::: "memory");
            __syncthreads();

            // ---- 4 WMMA K-steps over the staged block ----
#pragma unroll
            for (int ks = 0; ks < KBLK; ks += 32) {
                const int k = kb + ks;
                bf8v ah0 = *(const bf8v*)(fhiRow + k + hl * 16);
                bf8v ah1 = *(const bf8v*)(fhiRow + k + hl * 16 + 8);
                bf8v al0 = *(const bf8v*)(floRow + k + hl * 16);
                bf8v al1 = *(const bf8v*)(floRow + k + hl * 16 + 8);
                bf16v ahi = SHUF16(ah0, ah1);
                bf16v alo = SHUF16(al0, al1);
#define DO_TILE(T, ACC)                                                              \
                {                                                                    \
                    const unsigned short* bh = &sBhi[(T) * 16 + ln][ks + hl * 16];   \
                    const unsigned short* bl = &sBlo[(T) * 16 + ln][ks + hl * 16];   \
                    bf16v bhi = SHUF16(*(const bf8v*)bh, *(const bf8v*)(bh + 8));    \
                    bf16v blo = SHUF16(*(const bf8v*)bl, *(const bf8v*)(bl + 8));    \
                    ACC = WMMA_BF16(ahi, bhi, ACC);                                  \
                    ACC = WMMA_BF16(ahi, blo, ACC);                                  \
                    ACC = WMMA_BF16(alo, bhi, ACC);                                  \
                }
                DO_TILE(0, acc0)
                DO_TILE(1, acc1)
                DO_TILE(2, acc2)
                DO_TILE(3, acc3)
#undef DO_TILE
            }
            __syncthreads();   // protect staged B before next block overwrites it
        }

        // ---- distances for the 4 tiles of this column group ----
        float m2v[CTILES];
#pragma unroll
        for (int t = 0; t < CTILES; ++t) m2v[t] = m2[colBase + t * 16 + ln];

#pragma unroll
        for (int r = 0; r < 8; ++r) {
            float d0 = f2v[r] + m2v[0] - 2.0f * acc0[r];
            float d1 = f2v[r] + m2v[1] - 2.0f * acc1[r];
            float d2 = f2v[r] + m2v[2] - 2.0f * acc2[r];
            float d3 = f2v[r] + m2v[3] - 2.0f * acc3[r];
            sDist[wv][0][r + hl * 8][ln] = sqrtf(fmaxf(d0, 0.0f));
            sDist[wv][1][r + hl * 8][ln] = sqrtf(fmaxf(d1, 0.0f));
            sDist[wv][2][r + hl * 8][ln] = sqrtf(fmaxf(d2, 0.0f));
            sDist[wv][3][r + hl * 8][ln] = sqrtf(fmaxf(d3, 0.0f));
        }
        __syncthreads();

        if (lane < 16) {
#pragma unroll
            for (int t = 0; t < CTILES; ++t) {
#pragma unroll
                for (int nn = 0; nn < 16; ++nn) {
                    float dd = sDist[wv][t][lane][nn];
                    int   ii = colBase + t * 16 + nn;
                    if (dd < bestd[TOPK - 1]) {
#pragma unroll
                        for (int j = 0; j < TOPK; ++j) {
                            if (dd < bestd[j]) {
                                float td = bestd[j]; int ti = besti[j];
                                bestd[j] = dd; besti[j] = ii;
                                dd = td; ii = ti;
                            }
                        }
                    }
                }
            }
        }
        __syncthreads();
    }

    if (lane < 16) {
        const int rowTile16 = blockIdx.x * (QTILE / 16) + wv;
        size_t base = (((size_t)rowTile16 * NSLICE + blockIdx.y) * 16 + lane) * TOPK;
#pragma unroll
        for (int j = 0; j < TOPK; ++j) { pkd[base + j] = bestd[j]; pki[base + j] = besti[j]; }
    }
}

// ---------------- K5: merge 8 partial sorted top-9 lists per row ----------------
__global__ __launch_bounds__(256) void k_merge(const float* __restrict__ pkd,
                                               const int* __restrict__ pki,
                                               float* __restrict__ ds,
                                               int* __restrict__ idx9) {
    int n = blockIdx.x * 256 + threadIdx.x;
    if (n >= NQ) return;
    int rowTile = n >> 4; int m = n & 15;
    float bd[TOPK]; int bi[TOPK];
#pragma unroll
    for (int j = 0; j < TOPK; ++j) { bd[j] = 3.0e38f; bi[j] = 0; }
    for (int s = 0; s < NSLICE; ++s) {
        size_t base = (((size_t)rowTile * NSLICE + s) * 16 + m) * TOPK;
        for (int j = 0; j < TOPK; ++j) {
            float dd = pkd[base + j];
            if (!(dd < bd[TOPK - 1])) break;  // partial lists are sorted ascending
            int ii = pki[base + j];
#pragma unroll
            for (int q = 0; q < TOPK; ++q) {
                if (dd < bd[q]) {
                    float td = bd[q]; int ti = bi[q];
                    bd[q] = dd; bi[q] = ii;
                    dd = td; ii = ti;
                }
            }
        }
    }
    float s9 = 0.f;
#pragma unroll
    for (int j = 0; j < TOPK; ++j) { s9 += bd[j]; idx9[(size_t)n * TOPK + j] = bi[j]; }
    ds[n] = s9 * (1.0f / TOPK);
}

// ---------------- K6: global mean/std of distance signal (ddof=1 over N*C broadcast) ----------------
__global__ __launch_bounds__(256) void k_stats(const float* __restrict__ ds,
                                               float* __restrict__ stats) {
    double s1 = 0.0, s2 = 0.0;
    for (int n = threadIdx.x; n < NQ; n += 256) {
        double v = (double)ds[n]; s1 += v; s2 += v * v;
    }
    __shared__ double r1[256], r2[256];
    r1[threadIdx.x] = s1; r2[threadIdx.x] = s2; __syncthreads();
    for (int st = 128; st > 0; st >>= 1) {
        if (threadIdx.x < st) { r1[threadIdx.x] += r1[threadIdx.x + st]; r2[threadIdx.x] += r2[threadIdx.x + st]; }
        __syncthreads();
    }
    if (threadIdx.x == 0) {
        double S1 = r1[0], S2 = r2[0];
        double mean = S1 / (double)NQ;
        double ss   = S2 - S1 * S1 / (double)NQ;          // sum of sq deviations over N
        double var  = ((double)CC * ss) / ((double)NQ * CC - 1.0);
        stats[0] = (float)mean;
        stats[1] = (float)sqrt(var > 0.0 ? var : 0.0);
    }
}

// ---------------- K7: gather + influence + normalize + sigmoid noise + outputs ----------------
__device__ inline float blockSum(float v, float* red) {
    int tid = threadIdx.x;
    red[tid] = v; __syncthreads();
    for (int st = 128; st > 0; st >>= 1) {
        if (tid < st) red[tid] += red[tid + st];
        __syncthreads();
    }
    float r = red[0]; __syncthreads();
    return r;
}

__global__ __launch_bounds__(256) void k_final(const float* __restrict__ feat,
                                               const float* __restrict__ mb,
                                               const float* __restrict__ iw,
                                               const float* __restrict__ dw,
                                               const float* __restrict__ noise,
                                               const int* __restrict__ idx9,
                                               const float* __restrict__ ds,
                                               const float* __restrict__ stats,
                                               float* __restrict__ out) {
    int n = blockIdx.x;
    int b = n / HWW; int rem = n % HWW;
    size_t hw = (size_t)rem;
    int tid = threadIdx.x;

    __shared__ float fs[CC];
    __shared__ float red[256];
    __shared__ int nbr[TOPK];

    for (int c = tid; c < CC; c += 256) fs[c] = feat[((size_t)b * CC + c) * HWW + hw];
    if (tid < TOPK) nbr[tid] = idx9[(size_t)n * TOPK + tid];
    __syncthreads();

    float accv[8];
#pragma unroll
    for (int u = 0; u < 8; ++u) accv[u] = 0.f;
    for (int j = 0; j < TOPK; ++j) {
        const float* mrow = mb + (size_t)nbr[j] * CC;
#pragma unroll
        for (int u = 0; u < 8; ++u) {
            int c = tid + u * 256;
            accv[u] += fabsf(fs[c] - mrow[c]);
        }
    }

    float score[8];
    float s1 = 0.f, s2 = 0.f;
#pragma unroll
    for (int u = 0; u < 8; ++u) {
        int c = tid + u * 256;
        float sc = (accv[u] * (1.0f / TOPK)) * iw[c];
        score[u] = sc; s1 += sc; s2 += sc * sc;
    }
    float S1 = blockSum(s1, red);
    float S2 = blockSum(s2, red);
    float mean = S1 / (float)CC;
    float var  = (S2 - S1 * S1 / (float)CC) / (float)(CC - 1);
    float istd = sqrtf(fmaxf(var, 0.f));

    float dnorm = (ds[n] - stats[0]) / (stats[1] + EPSV);
    float dwv   = dw[0];

    float snsum = 0.f;
#pragma unroll
    for (int u = 0; u < 8; ++u) {
        int c = tid + u * 256;
        float inorm = (score[u] - mean) / (istd + EPSV);
        float comb  = inorm + dwv * dnorm;
        float sg    = 1.0f / (1.0f + expf(-comb));
        float nstd  = 0.01f + 0.49f * sg;
        snsum += nstd;
        out[((size_t)b * CC + c) * HWW + hw] = fs[c] + noise[(size_t)n * CC + c] * nstd;
    }
    float SN = blockSum(snsum, red);
    if (tid == 0) {
        const size_t mapBase = (size_t)BB * CC * HWW;
        out[mapBase + n]      = S1 / (float)CC;  // influence_map
        out[mapBase + NQ + n] = SN / (float)CC;  // noise_std_map
    }
}

// ---------------- host launcher ----------------
extern "C" void kernel_launch(void* const* d_in, const int* in_sizes, int n_in,
                              void* d_out, int out_size, void* d_ws, size_t ws_size,
                              hipStream_t stream) {
    const float* features = (const float*)d_in[0];
    const float* mb       = (const float*)d_in[1];
    const float* iw       = (const float*)d_in[2];
    const float* dw       = (const float*)d_in[3];
    const float* noise    = (const float*)d_in[4];
    float* out = (float*)d_out;

    char* ws = (char*)d_ws;
    size_t off = 0;
    auto take = [&](size_t bytes) -> char* {
        char* p = ws + off;
        off = (off + bytes + 255) & ~(size_t)255;
        return p;
    };
    unsigned short* mhi = (unsigned short*)take((size_t)MM * CC * 2);
    unsigned short* mlo = (unsigned short*)take((size_t)MM * CC * 2);
    float*          m2  = (float*)take((size_t)MM * 4);
    unsigned short* fhi = (unsigned short*)take((size_t)NQ * CC * 2);
    unsigned short* flo = (unsigned short*)take((size_t)NQ * CC * 2);
    float*          f2  = (float*)take((size_t)NQ * 4);
    float*          pkd = (float*)take((size_t)NROWT * NSLICE * 16 * TOPK * 4);
    int*            pki = (int*)take((size_t)NROWT * NSLICE * 16 * TOPK * 4);
    float*          dsv = (float*)take((size_t)NQ * 4);
    int*            idx9 = (int*)take((size_t)NQ * TOPK * 4);
    float*          stats = (float*)take(64);
    (void)ws_size; (void)in_sizes; (void)n_in; (void)out_size;

    k_split<<<((size_t)MM * CC + 255) / 256, 256, 0, stream>>>(mb, mhi, mlo, MM * CC);
    k_rownorm2<<<MM, 256, 0, stream>>>(mb, m2);
    k_convf<<<NQ, 256, 0, stream>>>(features, fhi, flo, f2);
    dim3 g4(NQ / QTILE, NSLICE);
    k_gemm_topk<<<g4, 128, 0, stream>>>(fhi, flo, mhi, mlo, f2, m2, pkd, pki);
    k_merge<<<(NQ + 255) / 256, 256, 0, stream>>>(pkd, pki, dsv, idx9);
    k_stats<<<1, 256, 0, stream>>>(dsv, stats);
    k_final<<<NQ, 256, 0, stream>>>(features, mb, iw, dw, noise, idx9, dsv, stats, out);
}
